// Bottleneck_refine_4715874091678
// MI455X (gfx1250) — compile-verified, matrix operand order checked by reading, not run on hardware
//
#include <hip/hip_runtime.h>

// ---------------------------------------------------------------------------
// Masked grouped bottleneck (SG-Net style) for MI455X / gfx1250.
// k0: BN scale/bias tables.  k1..k3: WMMA bf16 GEMMs with fused mask/BN/ReLU/
// residual epilogues.  Intermediates are bf16, channel-last per group
// [b][g][pix][32] so B-tile staging is contiguous -> async load-to-LDS where
// the toolchain exposes it (guarded), packed uint4 epilogue stores.
// ---------------------------------------------------------------------------

typedef __attribute__((ext_vector_type(16))) __bf16 v16bf;
typedef __attribute__((ext_vector_type(8)))  float  v8f;
typedef __attribute__((ext_vector_type(2)))  int    v2i;

union FragU { v16bf v; unsigned int u[8]; };

#define WMMA_BF16(A, B, C) \
    __builtin_amdgcn_wmma_f32_16x16x32_bf16(false, (A), false, (B), (short)0, (C), false, false)

__device__ __forceinline__ unsigned short f2bf(float x) {
    unsigned int u = __float_as_uint(x);
    u += 0x7FFFu + ((u >> 16) & 1u);        // round-to-nearest-even
    return (unsigned short)(u >> 16);
}

// Build one 16x32 bf16 WMMA operand fragment from LDS.
// `base` points at (row_or_col = lane&15, kbase + (lane>>4)*8).
// Per ISA layout: vgpr v<4 holds K pair {2v,2v+1}; v>=4 holds {16+2(v-4),...}.
__device__ __forceinline__ v16bf load_frag(const unsigned short* base) {
    FragU f;
#pragma unroll
    for (int v = 0; v < 8; ++v) {
        const int k = (v < 4) ? (v * 2) : (16 + (v - 4) * 2);
        f.u[v] = *(const unsigned int*)(base + k);   // 2 bf16 packed, 4B aligned
    }
    return f.v;
}

// ---- optional gfx1250 async global->LDS path (ASYNCcnt) -------------------
#if defined(__has_builtin)
#if __has_builtin(__builtin_amdgcn_global_load_async_to_lds_b64)
#define HAVE_ASYNC_LDS 1
#endif
#endif

#if defined(HAVE_ASYNC_LDS)
typedef __attribute__((address_space(1))) v2i gv2i;   // global int2
typedef __attribute__((address_space(3))) v2i lv2i;   // LDS int2
__device__ __forceinline__ void async_copy8(const void* g, void* l) {
    __builtin_amdgcn_global_load_async_to_lds_b64(
        (gv2i*)(unsigned long long)(uintptr_t)g,      // inttoptr -> AS1
        (lv2i*)l,                                     // addrspacecast -> AS3
        0, 0);
}
__device__ __forceinline__ void async_wait() {
#if __has_builtin(__builtin_amdgcn_s_wait_asynccnt)
    __builtin_amdgcn_s_wait_asynccnt(0);
#else
    asm volatile("s_wait_asynccnt 0x0" ::: "memory");
#endif
}
#else
__device__ __forceinline__ void async_copy8(const void* g, void* l) {
    *(unsigned long long*)l = *(const unsigned long long*)g;
}
__device__ __forceinline__ void async_wait() {}
#endif

// ---------------------------------------------------------------------------
// Kernel 0: per-channel BN tables.
// tab: [s1:0..127][bia1:128..255][s2:256..383][bia2:384..511]
//      [s3:512..1023][bia3:1024..1535]
// ---------------------------------------------------------------------------
__global__ void k0_prep(const float* __restrict__ g1, const float* __restrict__ b1,
                        const float* __restrict__ m1, const float* __restrict__ v1,
                        const float* __restrict__ g2, const float* __restrict__ b2,
                        const float* __restrict__ m2, const float* __restrict__ v2,
                        const float* __restrict__ g3, const float* __restrict__ b3,
                        const float* __restrict__ m3, const float* __restrict__ v3,
                        float* __restrict__ tab)
{
    const int t = threadIdx.x;
    if (t < 128) {
        float s = g1[t] * rsqrtf(v1[t] + 1e-5f);
        tab[t] = s;        tab[128 + t] = b1[t] - m1[t] * s;
        s = g2[t] * rsqrtf(v2[t] + 1e-5f);
        tab[256 + t] = s;  tab[384 + t] = b2[t] - m2[t] * s;
    }
    for (int c = t; c < 512; c += 256) {
        const float s = g3[c] * rsqrtf(v3[c] + 1e-5f);
        tab[512 + c] = s;  tab[1024 + c] = b3[c] - m3[c] * s;
    }
}

// ---------------------------------------------------------------------------
// Kernel 1: mid1 = mask * relu(s1*conv1x1_g4(mask*x) + bia1)   -> bf16
// Per block: (b, g, 112-pixel tile). GEMM [32 x 128] * [128 x 112].
// BN scale folded into A at load; mid1 layout [b][g][pix][32].
// ---------------------------------------------------------------------------
__global__ __launch_bounds__(224)
void k1_conv1(const float* __restrict__ x, const float* __restrict__ mask,
              const float* __restrict__ w1, const float* __restrict__ tab,
              unsigned short* __restrict__ mid1)
{
    const int tile = blockIdx.x, g = blockIdx.y, b = blockIdx.z;
    const int p0 = tile * 112;
    const int b4g = b * 4 + g;

    __shared__ __align__(16) unsigned short ldsA[32 * 130];   // [oc][ic]
    __shared__ __align__(16) unsigned short ldsB[112 * 132];  // [pix][ic]
    __shared__ float maskv[112];

    const int tid = threadIdx.x;
    const int lane = tid & 31, wave = tid >> 5;
    const int lr = lane & 15, hi = lane >> 4;

    if (tid < 112) {
        const int p = p0 + tid, h = p / 56, w = p % 56;
        maskv[tid] = mask[(b4g * 7 + (h >> 3)) * 7 + (w >> 3)];
    }
    for (int idx = tid; idx < 32 * 128; idx += 224) {
        const int oc = idx >> 7, ic = idx & 127;
        ldsA[oc * 130 + ic] = f2bf(w1[(g * 32 + oc) * 128 + ic] * tab[g * 32 + oc]);
    }
    __syncthreads();
    for (int idx = tid; idx < 128 * 112; idx += 224) {
        const int ic = idx / 112, pix = idx % 112;
        const float val = x[((size_t)b * 512 + g * 128 + ic) * 3136 + p0 + pix] * maskv[pix];
        ldsB[pix * 132 + ic] = f2bf(val);
    }
    __syncthreads();

    const int nt = wave;
    v8f acc[2]; const v8f z = {}; acc[0] = z; acc[1] = z;
#pragma unroll
    for (int kk = 0; kk < 4; ++kk) {
        const v16bf bf = load_frag(&ldsB[(nt * 16 + lr) * 132 + kk * 32 + hi * 8]);
#pragma unroll
        for (int mt = 0; mt < 2; ++mt) {
            const v16bf af = load_frag(&ldsA[(mt * 16 + lr) * 130 + kk * 32 + hi * 8]);
            acc[mt] = WMMA_BF16(af, bf, acc[mt]);
        }
    }

    const int n = nt * 16 + lr;
    const float mval = maskv[n];
    const size_t pixbase = ((size_t)b4g * 3136 + p0 + n) * 32;
#pragma unroll
    for (int mt = 0; mt < 2; ++mt) {
        unsigned int pk[4];
#pragma unroll
        for (int q = 0; q < 4; ++q) {
            const int c0 = g * 32 + mt * 16 + hi * 8 + 2 * q;
            const float e0 = fmaxf(acc[mt][2 * q]     + tab[128 + c0],     0.0f) * mval;
            const float e1 = fmaxf(acc[mt][2 * q + 1] + tab[128 + c0 + 1], 0.0f) * mval;
            pk[q] = (unsigned int)f2bf(e0) | ((unsigned int)f2bf(e1) << 16);
        }
        uint4 q4; q4.x = pk[0]; q4.y = pk[1]; q4.z = pk[2]; q4.w = pk[3];
        *(uint4*)&mid1[pixbase + mt * 16 + hi * 8] = q4;   // 16B packed store
    }
}

// ---------------------------------------------------------------------------
// Kernel 2: mid2 = mask * relu(s2*conv3x3_g4(mid1) + bia2)     -> bf16
// Per block: (b, g, row-pair). 9 tap-GEMMs [32 x 32] over haloed LDS tile.
// mid layout is channel-last: per-pixel 64B rows -> async copy to LDS.
// ---------------------------------------------------------------------------
__global__ __launch_bounds__(224)
void k2_conv2(const unsigned short* __restrict__ mid1, const float* __restrict__ mask,
              const float* __restrict__ w2, const float* __restrict__ tab,
              unsigned short* __restrict__ mid2)
{
    const int rp = blockIdx.x, g = blockIdx.y, b = blockIdx.z;
    const int h0 = rp * 2;
    const int b4g = b * 4 + g;

    __shared__ __align__(16) unsigned short ldsA[32 * 290];   // [oc][tap*32+ic]
    __shared__ __align__(16) unsigned short ldsB[232 * 36];   // [rr*58+cc][ic], 72B rows
    __shared__ float maskv[112];

    const int tid = threadIdx.x;
    const int lane = tid & 31, wave = tid >> 5;
    const int lr = lane & 15, hi = lane >> 4;

    if (tid < 112) {
        const int h = h0 + tid / 56, w = tid % 56;
        maskv[tid] = mask[(b4g * 7 + (h >> 3)) * 7 + (w >> 3)];
    }
    for (int idx = tid; idx < 32 * 288; idx += 224) {
        const int oc = idx / 288, rem = idx % 288;
        const int tap = rem >> 5, ic = rem & 31;
        ldsA[oc * 290 + rem] = f2bf(w2[((g * 32 + oc) * 32 + ic) * 9 + tap] * tab[256 + g * 32 + oc]);
    }
    // haloed tile: rows h0-1..h0+2, cols -1..56; 64B per in-range pixel
    for (int idx = tid; idx < 232 * 8; idx += 224) {
        const int pl = idx >> 3, part = idx & 7;
        const int rr = pl / 58, cc = pl % 58;
        const int h = h0 - 1 + rr, w = cc - 1;
        char* ldst = (char*)ldsB + pl * 72 + part * 8;
        if (h >= 0 && h < 56 && w >= 0 && w < 56) {
            const char* src = (const char*)mid1 + (((size_t)b4g * 3136 + h * 56 + w) * 32) * 2 + part * 8;
            async_copy8(src, ldst);
        } else {
            *(unsigned long long*)ldst = 0ull;
        }
    }
    async_wait();
    __syncthreads();

    const int nt = wave;
    const int n = nt * 16 + lr;
    const int orow = n / 56, ocol = n % 56;
    v8f acc[2]; const v8f z = {}; acc[0] = z; acc[1] = z;
#pragma unroll
    for (int tap = 0; tap < 9; ++tap) {
        const int dy = tap / 3, dx = tap % 3;
        const int pl = (orow + dy) * 58 + (ocol + dx);
        const v16bf bf = load_frag(&ldsB[pl * 36 + hi * 8]);
#pragma unroll
        for (int mt = 0; mt < 2; ++mt) {
            const v16bf af = load_frag(&ldsA[(mt * 16 + lr) * 290 + tap * 32 + hi * 8]);
            acc[mt] = WMMA_BF16(af, bf, acc[mt]);
        }
    }

    const float mval = maskv[n];
    const size_t pixbase = ((size_t)b4g * 3136 + h0 * 56 + n) * 32;
#pragma unroll
    for (int mt = 0; mt < 2; ++mt) {
        unsigned int pk[4];
#pragma unroll
        for (int q = 0; q < 4; ++q) {
            const int c0 = g * 32 + mt * 16 + hi * 8 + 2 * q;
            const float e0 = fmaxf(acc[mt][2 * q]     + tab[384 + c0],     0.0f) * mval;
            const float e1 = fmaxf(acc[mt][2 * q + 1] + tab[384 + c0 + 1], 0.0f) * mval;
            pk[q] = (unsigned int)f2bf(e0) | ((unsigned int)f2bf(e1) << 16);
        }
        uint4 q4; q4.x = pk[0]; q4.y = pk[1]; q4.z = pk[2]; q4.w = pk[3];
        *(uint4*)&mid2[pixbase + mt * 16 + hi * 8] = q4;
    }
}

// ---------------------------------------------------------------------------
// Kernel 3: out = relu(s3*conv1x1_g4(mid2) + bia3 + x)         -> f32
// Per block: (b, g, 112-pixel tile). GEMM [128 x 32] * [32 x 112].
// B tile is one contiguous 7168B global region -> async copy to LDS.
// ---------------------------------------------------------------------------
__global__ __launch_bounds__(224)
void k3_conv3(const unsigned short* __restrict__ mid2, const float* __restrict__ x,
              const float* __restrict__ w3, const float* __restrict__ tab,
              float* __restrict__ out)
{
    const int tile = blockIdx.x, g = blockIdx.y, b = blockIdx.z;
    const int p0 = tile * 112;
    const int b4g = b * 4 + g;

    __shared__ __align__(16) unsigned short ldsA[128 * 34];   // [oc][ic]
    __shared__ __align__(16) unsigned short ldsB[112 * 36];   // [pix][ic], 72B rows

    const int tid = threadIdx.x;
    const int lane = tid & 31, wave = tid >> 5;
    const int lr = lane & 15, hi = lane >> 4;

    // hint the residual into cache early (gfx1250 global_prefetch_b8)
    __builtin_prefetch(&x[((size_t)b * 512 + g * 128) * 3136 + p0 + tid], 0, 1);

    // B: contiguous 112*64B global -> padded LDS rows (async where available)
    const char* src_base = (const char*)mid2 + ((size_t)b4g * 3136 + p0) * 64;
    for (int idx = tid; idx < 112 * 8; idx += 224) {
        const int pl = idx >> 3, part = idx & 7;
        async_copy8(src_base + pl * 64 + part * 8, (char*)ldsB + pl * 72 + part * 8);
    }
    for (int idx = tid; idx < 128 * 32; idx += 224) {
        const int oc = idx >> 5, ic = idx & 31;
        ldsA[oc * 34 + ic] = f2bf(w3[(g * 128 + oc) * 32 + ic] * tab[512 + g * 128 + oc]);
    }
    async_wait();
    __syncthreads();

    const int nt = wave;
    const v16bf bf = load_frag(&ldsB[(nt * 16 + lr) * 36 + hi * 8]);
    v8f acc[8]; const v8f z = {};
#pragma unroll
    for (int mt = 0; mt < 8; ++mt) {
        const v16bf af = load_frag(&ldsA[(mt * 16 + lr) * 34 + hi * 8]);
        acc[mt] = WMMA_BF16(af, bf, z);
    }

    const int n = nt * 16 + lr;
    const size_t p = (size_t)(p0 + n);
#pragma unroll
    for (int mt = 0; mt < 8; ++mt) {
#pragma unroll
        for (int r = 0; r < 8; ++r) {
            const int oc = g * 128 + mt * 16 + hi * 8 + r;
            const size_t off = ((size_t)b * 512 + oc) * 3136 + p;
            out[off] = fmaxf(acc[mt][r] + tab[1024 + oc] + x[off], 0.0f);
        }
    }
}

// ---------------------------------------------------------------------------
extern "C" void kernel_launch(void* const* d_in, const int* in_sizes, int n_in,
                              void* d_out, int out_size, void* d_ws, size_t ws_size,
                              hipStream_t stream) {
    const float* x    = (const float*)d_in[0];
    const float* mask = (const float*)d_in[1];
    const float* w1 = (const float*)d_in[2];
    const float* g1 = (const float*)d_in[3];
    const float* b1 = (const float*)d_in[4];
    const float* m1 = (const float*)d_in[5];
    const float* v1 = (const float*)d_in[6];
    const float* w2 = (const float*)d_in[7];
    const float* g2 = (const float*)d_in[8];
    const float* b2 = (const float*)d_in[9];
    const float* m2 = (const float*)d_in[10];
    const float* v2 = (const float*)d_in[11];
    const float* w3 = (const float*)d_in[12];
    const float* g3 = (const float*)d_in[13];
    const float* b3 = (const float*)d_in[14];
    const float* m3 = (const float*)d_in[15];
    const float* v3 = (const float*)d_in[16];
    float* out = (float*)d_out;

    const size_t midElems = (size_t)16 * 128 * 3136;           // bf16 elements
    unsigned short* mid1 = (unsigned short*)d_ws;              // [16][4][3136][32]
    unsigned short* mid2 = mid1 + midElems;
    float* tab = (float*)(mid2 + midElems);                    // 1536 floats

    const dim3 blk(224);
    const dim3 grid(28, 4, 16);  // (pixel-tile / row-pair, group, batch)

    k0_prep<<<dim3(1), dim3(256), 0, stream>>>(g1, b1, m1, v1, g2, b2, m2, v2,
                                               g3, b3, m3, v3, tab);
    k1_conv1<<<grid, blk, 0, stream>>>(x, mask, w1, tab, mid1);
    k2_conv2<<<grid, blk, 0, stream>>>(mid1, mask, w2, tab, mid2);
    k3_conv3<<<grid, blk, 0, stream>>>(mid2, x, w3, tab, out);
}
